// FullAttention_82781199663274
// MI455X (gfx1250) — matrix operand order repaired and checked by reading, not compiled, not run on hardware
//
#include <hip/hip_runtime.h>
#include <math.h>

typedef __attribute__((ext_vector_type(16))) _Float16 v16h;
typedef __attribute__((ext_vector_type(8)))  _Float16 v8h;
typedef __attribute__((ext_vector_type(8)))  float    v8f;

// Problem constants (match reference setup_inputs)
#define B_   4
#define L_   2048
#define H_   8
#define E_   64

#define WAVES      8
#define ROWS_BLK   128    // rows per workgroup (8 waves x 16 rows)
#define KBLK       32     // keys per flash iteration (= WMMA f16 K dim)

// LDS padded strides (halves); all operand reads are contiguous & 16B aligned
#define KSTR  72          // K tile row stride   (64 + 8 pad) -> 144B rows
#define VSTR  40          // V^T tile row stride (32 + 8 pad) ->  80B rows
#define PSTR  40          // P tile row stride

// softmax run in base-2: 2^(c*x) == e^(scale*x) with c = scale*log2(e)
#define SCALE_LOG2E (0.125f * 1.4426950408889634f)

// ---------------------------------------------------------------------------
// One-time K/V precision+layout pass:
//   Kh  [bh][l][e]  f16   (row-major per (b,h); key tiles are contiguous 4KB)
//   Vth [bh][e][l]  f16   (transposed; V^T tiles are 64 rows x 64B)
// ---------------------------------------------------------------------------
__global__ __launch_bounds__(256) void convert_kv(const float* __restrict__ K,
                                                  const float* __restrict__ V,
                                                  _Float16* __restrict__ Kh,
                                                  _Float16* __restrict__ Vth)
{
    __shared__ __align__(16) _Float16 t[64 * 65];
    const int bh = blockIdx.y;
    const int b  = bh / H_, h = bh % H_;
    const int l0 = blockIdx.x * 64;
    const size_t base = ((size_t)b * L_ * H_ + h) * E_;
    const int tid = threadIdx.x;
    const int lr = tid >> 2;             // 0..63
    const int e0 = (tid & 3) * 16;       // 0,16,32,48

    // K: straight convert/copy (coalesced both sides)
    {
        const float* kp = K + base + (size_t)(l0 + lr) * (H_ * E_) + e0;
        _Float16*    ko = Kh + ((size_t)bh * L_ + l0 + lr) * E_ + e0;
        v8h a, c;
        #pragma unroll
        for (int i = 0; i < 8; ++i) { a[i] = (_Float16)kp[i]; c[i] = (_Float16)kp[8 + i]; }
        *(v8h*)&ko[0] = a; *(v8h*)&ko[8] = c;
    }
    // V: convert into LDS tile, write transposed (coalesced)
    {
        const float* vp = V + base + (size_t)(l0 + lr) * (H_ * E_) + e0;
        #pragma unroll
        for (int i = 0; i < 16; ++i) t[lr * 65 + e0 + i] = (_Float16)vp[i];
    }
    __syncthreads();
    {
        const int er = tid >> 2;             // e row 0..63
        const int p  = (tid & 3) * 16;       // l chunk
        _Float16* vo = Vth + ((size_t)bh * E_ + er) * L_ + l0 + p;
        v8h a, c;
        #pragma unroll
        for (int i = 0; i < 8; ++i) { a[i] = t[(p + i) * 65 + er]; c[i] = t[(p + 8 + i) * 65 + er]; }
        *(v8h*)&vo[0] = a; *(v8h*)&vo[8] = c;
    }
}

// ---------------------------------------------------------------------------
// Flash attention. F16SRC=true : stage pre-converted f16 K / V^T tiles with
// global_load_async_to_lds_b128 (ASYNCcnt), double-buffered + pipelined.
// F16SRC=false: fallback, convert fp32->f16 while staging (plain loads).
// ---------------------------------------------------------------------------
template <bool F16SRC>
__global__ __launch_bounds__(256) void
fa_causal(const float* __restrict__ Qf,
          const float* __restrict__ Kf, const float* __restrict__ Vf,
          const _Float16* __restrict__ Kh, const _Float16* __restrict__ Vth,
          float* __restrict__ Out)
{
    const int tid   = threadIdx.x;
    const int wave  = tid >> 5;
    const int lane  = tid & 31;
    const int half_ = lane >> 4;
    const int ln    = lane & 15;

    const int bh  = blockIdx.y;
    const int b   = bh / H_;
    const int h   = bh % H_;
    const int r0b = blockIdx.x * ROWS_BLK;
    const int r0w = r0b + wave * 16;

    __shared__ __align__(16) _Float16 ldsK[2][KBLK * KSTR];       // [buf][key][e]
    __shared__ __align__(16) _Float16 ldsV[2][E_ * VSTR];         // [buf][e][key]
    __shared__ __align__(16) _Float16 ldsP[WAVES * 16 * PSTR];    // per-wave [m][k]
    _Float16* myP = &ldsP[wave * 16 * PSTR];

    const size_t base  = ((size_t)b * L_ * H_ + h) * E_;
    const int rowStr   = H_ * E_;

    // ---- Q strip as two f16 A-operands, scale*log2e folded in ----
    v16h aQ[2];
    {
        const float* qrow = Qf + base + (size_t)(r0w + ln) * rowStr;
        #pragma unroll
        for (int c = 0; c < 2; ++c) {
            const int e0 = c * 32 + half_ * 16;
            #pragma unroll
            for (int j = 0; j < 8; ++j) {
                aQ[c][j]     = (_Float16)(qrow[e0 + j]      * SCALE_LOG2E);
                aQ[c][j + 8] = (_Float16)(qrow[e0 + 8 + j] * SCALE_LOG2E);
            }
        }
    }

    v8f  o[4] = {};
    float Mrow[8], Lrow[8];
    #pragma unroll
    for (int r = 0; r < 8; ++r) { Mrow[r] = -__builtin_inff(); Lrow[r] = 0.f; }

    const int nIter = (r0b + ROWS_BLK) / KBLK;

    // ---- tile staging (whole workgroup) ----
    auto stage = [&](int it, int buf) {
        const int kb = it * KBLK;
        if constexpr (F16SRC) {
            // K tile: 32 keys x 128B, contiguous in Kh; 16B per thread
            {
                const int kr = tid >> 3, part = tid & 7;
                unsigned long long ga = (unsigned long long)(uintptr_t)
                    (Kh + ((size_t)bh * L_ + kb + kr) * E_ + part * 8);
                unsigned int la = (unsigned int)(uintptr_t)&ldsK[buf][kr * KSTR + part * 8];
                asm volatile("global_load_async_to_lds_b128 %0, %1, off"
                             :: "v"(la), "v"(ga) : "memory");
            }
            // V^T tile: 64 e-rows x 64B from Vth; 16B per thread
            {
                const int er = tid >> 2, p2 = tid & 3;
                unsigned long long ga = (unsigned long long)(uintptr_t)
                    (Vth + ((size_t)bh * E_ + er) * L_ + kb + p2 * 8);
                unsigned int la = (unsigned int)(uintptr_t)&ldsV[buf][er * VSTR + p2 * 8];
                asm volatile("global_load_async_to_lds_b128 %0, %1, off"
                             :: "v"(la), "v"(ga) : "memory");
            }
        } else {
            const int  kr = tid >> 3;
            const int  e0 = (tid & 7) * 8;
            const float* kp = Kf + base + (size_t)(kb + kr) * rowStr + e0;
            const float* vp = Vf + base + (size_t)(kb + kr) * rowStr + e0;
            v8h khv, vhv;
            #pragma unroll
            for (int i = 0; i < 8; ++i) { khv[i] = (_Float16)kp[i]; vhv[i] = (_Float16)vp[i]; }
            *(v8h*)&ldsK[buf][kr * KSTR + e0] = khv;
            #pragma unroll
            for (int i = 0; i < 8; ++i) ldsV[buf][(e0 + i) * VSTR + kr] = vhv[i];
        }
    };

    stage(0, 0);   // prologue

    for (int i = 0; i < nIter; ++i) {
        if constexpr (F16SRC)
            asm volatile("s_wait_asynccnt 0x0" ::: "memory");  // tile i landed
        __syncthreads();                                        // visible to all
        if (i + 1 < nIter) stage(i + 1, (i + 1) & 1);           // overlap next tile

        const int kb = i * KBLK;
        if (kb > r0w + 15) continue;        // fully masked for this wave

        const _Float16* lK = ldsK[i & 1];
        const _Float16* lV = ldsV[i & 1];

        // ---- S = (scaled Q) K^T : two 16x16 tiles, K-dim 64 = 2 chunks ----
        v8f s[2];
        #pragma unroll
        for (int t = 0; t < 2; ++t) {
            v8f c = {};
            #pragma unroll
            for (int cc = 0; cc < 2; ++cc) {
                v16h bK = *(const v16h*)&lK[(t * 16 + ln) * KSTR + cc * 32 + half_ * 16];
                c = __builtin_amdgcn_wmma_f32_16x16x32_f16(false, aQ[cc], false, bK,
                                                           (short)0, c, false, false);
            }
            s[t] = c;
        }

        // ---- causal mask only on the (single) diagonal block ----
        // NOTE: the reference's ALiBi term triu(slope*(k-q)) is non-zero only
        // where k>q, which the causal mask sends to -inf; at k==q it is 0.
        // Net contribution is exactly zero, so it is omitted.
        float bmax[8];
        if (kb + KBLK - 1 > r0w) {          // wave-uniform
            #pragma unroll
            for (int r = 0; r < 8; ++r) {
                const int q = r0w + r + 8 * half_;
                #pragma unroll
                for (int t = 0; t < 2; ++t)
                    if (kb + t * 16 + ln > q) s[t][r] = -__builtin_inff();
                bmax[r] = fmaxf(s[0][r], s[1][r]);
            }
        } else {
            #pragma unroll
            for (int r = 0; r < 8; ++r) bmax[r] = fmaxf(s[0][r], s[1][r]);
        }
        #pragma unroll
        for (int off = 1; off < 16; off <<= 1)
            #pragma unroll
            for (int r = 0; r < 8; ++r)
                bmax[r] = fmaxf(bmax[r], __shfl_xor(bmax[r], off, 32));

        // ---- online softmax (base-2 domain) ----
        float alpha[8], rsum[8];
        #pragma unroll
        for (int r = 0; r < 8; ++r) {
            const float mn = fmaxf(Mrow[r], bmax[r]);
            alpha[r] = __builtin_exp2f(Mrow[r] - mn);
            Mrow[r]  = mn;
        }
        #pragma unroll
        for (int r = 0; r < 8; ++r) {
            float acc = 0.f;
            #pragma unroll
            for (int t = 0; t < 2; ++t) {
                const float p = __builtin_exp2f(s[t][r] - Mrow[r]);
                s[t][r] = p;
                acc += p;
            }
            rsum[r] = acc;
        }
        #pragma unroll
        for (int off = 1; off < 16; off <<= 1)
            #pragma unroll
            for (int r = 0; r < 8; ++r)
                rsum[r] += __shfl_xor(rsum[r], off, 32);
        #pragma unroll
        for (int r = 0; r < 8; ++r)
            Lrow[r] = Lrow[r] * alpha[r] + rsum[r];
        #pragma unroll
        for (int nc = 0; nc < 4; ++nc)
            #pragma unroll
            for (int r = 0; r < 8; ++r)
                o[nc][r] *= alpha[r];

        // ---- P (C/D layout) -> per-wave LDS -> A-operand layout ----
        #pragma unroll
        for (int t = 0; t < 2; ++t)
            #pragma unroll
            for (int r = 0; r < 8; ++r)
                myP[(r + 8 * half_) * PSTR + t * 16 + ln] = (_Float16)s[t][r];
        __builtin_amdgcn_wave_barrier();   // DS in-order per wave; stop reordering

        v16h aP;
        {
            const _Float16* pr = &myP[ln * PSTR];
            const v8h lo = *(const v8h*)&pr[half_ * 16];
            const v8h hi = *(const v8h*)&pr[half_ * 16 + 8];
            #pragma unroll
            for (int j = 0; j < 8; ++j) { aP[j] = lo[j]; aP[j + 8] = hi[j]; }
        }

        // ---- O += P * V : 4 n-chunks from transposed-V LDS ----
        #pragma unroll
        for (int nc = 0; nc < 4; ++nc) {
            v16h bV = *(const v16h*)&lV[(nc * 16 + ln) * VSTR + half_ * 16];
            o[nc] = __builtin_amdgcn_wmma_f32_16x16x32_f16(false, aP, false, bV,
                                                           (short)0, o[nc], false, false);
        }
    }

    // ---- normalize and store [B,L,H,E] ----
    #pragma unroll
    for (int r = 0; r < 8; ++r) {
        const float inv = 1.0f / Lrow[r];
        const int   m   = r + 8 * half_;
        float* orow = Out + ((size_t)(b * L_ + r0w + m) * H_ + h) * E_;
        #pragma unroll
        for (int nc = 0; nc < 4; ++nc)
            orow[nc * 16 + ln] = o[nc][r] * inv;
    }
}

extern "C" void kernel_launch(void* const* d_in, const int* in_sizes, int n_in,
                              void* d_out, int out_size, void* d_ws, size_t ws_size,
                              hipStream_t stream) {
    const float* Q = (const float*)d_in[0];
    const float* K = (const float*)d_in[1];
    const float* V = (const float*)d_in[2];
    // d_in[3] = attn_mask (bool causal) -- computed analytically.
    float* Out = (float*)d_out;

    const size_t nElem = (size_t)B_ * H_ * L_ * E_;
    const size_t need  = 2 * nElem * sizeof(_Float16);

    dim3 agrid(L_ / ROWS_BLK, B_ * H_);
    dim3 ablock(256);

    if (ws_size >= need) {
        _Float16* Kh  = (_Float16*)d_ws;
        _Float16* Vth = Kh + nElem;
        hipLaunchKernelGGL(convert_kv, dim3(L_ / 64, B_ * H_), dim3(256), 0, stream,
                           K, V, Kh, Vth);
        hipLaunchKernelGGL((fa_causal<true>), agrid, ablock, 0, stream,
                           Q, K, V, Kh, Vth, Out);
    } else {
        hipLaunchKernelGGL((fa_causal<false>), agrid, ablock, 0, stream,
                           Q, K, V, (const _Float16*)nullptr, (const _Float16*)nullptr, Out);
    }
}
